// LWAttention_89429809037486
// MI455X (gfx1250) — matrix operand order repaired and checked by reading, not compiled, not run on hardware
//
#include <hip/hip_runtime.h>
#include <cstdint>

typedef __bf16 bf16;
typedef __attribute__((ext_vector_type(16))) __bf16 v16bf;
typedef __attribute__((ext_vector_type(8)))  __bf16 v8bf;
typedef __attribute__((ext_vector_type(8)))  float  v8f;

#define IN_CH  256
#define QK     64
#define OUT_CH 256
#define N_SP   4096
#define BATCH  8

// LDS strides (elements) chosen so ds_load_b128 bank steps cycle all residues
#define KS_STRIDE 72    // 144B per 32-key row (128B data + pad)
#define VS_STRIDE 40    // 80B per V^T row (64B data + pad)
#define KS_BUF (32 * KS_STRIDE)
#define VS_BUF (256 * VS_STRIDE)

static __device__ __forceinline__ v8f wmma_bf16(v16bf a, v16bf b, v8f c) {
  // D = A(16x32 bf16) * B(32x16 bf16) + C(16x16 f32)
  return __builtin_amdgcn_wmma_f32_16x16x32_bf16(false, a, false, b, (short)0, c,
                                                 false, false);
}

// SWAPX16: exchange lane L <-> lane L^16 (group-of-32 swizzle, xor=0x10, and=0x1f)
static __device__ __forceinline__ float swz16f(float v) {
  return __int_as_float(__builtin_amdgcn_ds_swizzle(__float_as_int(v), 0x401f));
}
static __device__ __forceinline__ int swz16i(int v) {
  return __builtin_amdgcn_ds_swizzle(v, 0x401f);
}

// One 16B-per-lane async global->LDS copy (ASYNCcnt-tracked, GV addressing).
static __device__ __forceinline__ void async_copy16(unsigned lds_off, const void* g) {
  asm volatile("global_load_async_to_lds_b128 %0, %1, off"
               :: "v"(lds_off), "v"((unsigned long long)(uintptr_t)g)
               : "memory");
}
static __device__ __forceinline__ void wait_async0() {
  asm volatile("s_wait_asynccnt 0" ::: "memory");
}

// ---------------------------------------------------------------------------
// Kernel 1: 1x1-conv projections  Q,K,V = W * x + b   (WMMA bf16, f32 acc)
//   x  : [B, 256, 4096] f32 (c-major)
//   Qw : [B, 4096, 64]  bf16   Kw : [B, 4096, 64] bf16
//   Vt : [B, 256, 4096] bf16  (Vt[d][m] = V[m][d])
// ---------------------------------------------------------------------------
__global__ __launch_bounds__(256) void proj_kernel(
    const float* __restrict__ x,
    const float* __restrict__ w_q, const float* __restrict__ b_q,
    const float* __restrict__ w_k, const float* __restrict__ b_k,
    const float* __restrict__ w_v, const float* __restrict__ b_v,
    bf16* __restrict__ Qw, bf16* __restrict__ Kw, bf16* __restrict__ Vt) {
  __shared__ __align__(32) bf16 xs[128 * 272];  // [n][c], 32B-aligned rows

  const int b  = blockIdx.y;
  const int n0 = blockIdx.x * 128;
  const int tid = threadIdx.x;

  const float* xb = x + (size_t)b * IN_CH * N_SP + n0;
  for (int i = tid; i < 256 * 32; i += 256) {
    const int c = i >> 5, g = (i & 31) << 2;
    const float4 f = *reinterpret_cast<const float4*>(xb + (size_t)c * N_SP + g);
    xs[(g + 0) * 272 + c] = (bf16)f.x;
    xs[(g + 1) * 272 + c] = (bf16)f.y;
    xs[(g + 2) * 272 + c] = (bf16)f.z;
    xs[(g + 3) * 272 + c] = (bf16)f.w;
  }
  __syncthreads();

  const int wave = tid >> 5, lane = tid & 31;
  const int col = lane & 15, half = lane >> 4;

  for (int t = wave; t < 192; t += 8) {
    const int dt = t >> 3, nt = t & 7;
    const float* W; const float* bias; int dbase;
    if (dt < 4)      { W = w_q; bias = b_q; dbase = dt * 16; }
    else if (dt < 8) { W = w_k; bias = b_k; dbase = (dt - 4) * 16; }
    else             { W = w_v; bias = b_v; dbase = (dt - 8) * 16; }

    const float* wrow = W + (size_t)(dbase + col) * IN_CH;
    const bf16*  xrow = &xs[(nt * 16 + col) * 272];
    v8f acc;
#pragma unroll
    for (int r = 0; r < 8; ++r) acc[r] = 0.f;

#pragma unroll
    for (int c0 = 0; c0 < 256; c0 += 32) {
      const v8f f0 = *reinterpret_cast<const v8f*>(wrow + c0 + half * 16);
      const v8f f1 = *reinterpret_cast<const v8f*>(wrow + c0 + half * 16 + 8);
      union { v16bf v; bf16 e[16]; } A;
#pragma unroll
      for (int j = 0; j < 8; ++j) { A.e[j] = (bf16)f0[j]; A.e[8 + j] = (bf16)f1[j]; }
      const v16bf B = *reinterpret_cast<const v16bf*>(xrow + c0 + half * 16);
      acc = wmma_bf16(A.v, B, acc);
    }

#pragma unroll
    for (int r = 0; r < 8; ++r) acc[r] += bias[dbase + half * 8 + r];

    const int gn = n0 + nt * 16 + col;
    if (dt < 8) {
      bf16* dst = (dt < 4 ? Qw : Kw) + ((size_t)b * N_SP + gn) * QK + dbase + half * 8;
      v8bf pk;
#pragma unroll
      for (int r = 0; r < 8; ++r) pk[r] = (bf16)acc[r];
      *reinterpret_cast<v8bf*>(dst) = pk;
    } else {
      bf16* dst = Vt + (size_t)b * OUT_CH * N_SP + gn;
#pragma unroll
      for (int r = 0; r < 8; ++r)
        dst[(size_t)(dbase + half * 8 + r) * N_SP] = (bf16)acc[r];
    }
  }
}

// ---------------------------------------------------------------------------
// Kernel 2: flash attention, 4 waves/block sharing async-staged K/V chunks.
// Wave w owns query tile n0 = blockIdx.x*64 + w*16; 128 chunks of 32 keys.
// K chunk (32x64 bf16) + V^T chunk (256x32 bf16) double-buffered in LDS via
// global_load_async_to_lds_b128; s_wait_asynccnt + barrier per chunk.
// ---------------------------------------------------------------------------
__global__ __launch_bounds__(128) void attn_kernel(
    const bf16* __restrict__ Qw, const bf16* __restrict__ Kw,
    const bf16* __restrict__ Vt, float* __restrict__ out) {
  __shared__ __align__(16) bf16 Ks[2 * KS_BUF];   //  9.2 KB
  __shared__ __align__(16) bf16 Vs[2 * VS_BUF];   // 40.9 KB

  const int b   = blockIdx.y;
  const int tid = threadIdx.x;
  const int wave = tid >> 5, lane = tid & 31;
  const int col = lane & 15, half = lane >> 4;
  const int n0  = blockIdx.x * 64 + wave * 16;

  const bf16* Kb = Kw + (size_t)b * N_SP * QK;
  const bf16* Vb = Vt + (size_t)b * OUT_CH * N_SP;

  // Q^T B-operands: lane n holds reduction channels c = half*16 .. +15 (+32)
  const bf16* qrow = Qw + ((size_t)b * N_SP + n0 + col) * QK;
  const v16bf bq0 = *reinterpret_cast<const v16bf*>(qrow + half * 16);
  const v16bf bq1 = *reinterpret_cast<const v16bf*>(qrow + 32 + half * 16);

  v8f acc[16];
#pragma unroll
  for (int t = 0; t < 16; ++t)
#pragma unroll
    for (int r = 0; r < 8; ++r) acc[t][r] = 0.f;

  float mi = -3.0e38f, li = 0.f;

  // ---- async stage of one 32-key chunk into LDS buffer `buf` ----
  auto stage = [&](int m0, int buf) {
    bf16* VsB = Vs + buf * VS_BUF;
    bf16* KsB = Ks + buf * KS_BUF;
#pragma unroll
    for (int j = 0; j < 8; ++j) {            // V^T: 256 rows x 64B
      const int e = j * 128 + tid;
      const int d = e >> 2, off = (e & 3) * 8;   // off in elements (16B units)
      async_copy16((unsigned)(uintptr_t)(VsB + d * VS_STRIDE + off),
                   Vb + (size_t)d * N_SP + m0 + off);
    }
#pragma unroll
    for (int j = 0; j < 2; ++j) {            // K: 32 rows x 128B
      const int e = j * 128 + tid;
      const int m = e >> 3, off = (e & 7) * 8;
      async_copy16((unsigned)(uintptr_t)(KsB + m * KS_STRIDE + off),
                   Kb + (size_t)(m0 + m) * QK + off);
    }
  };

  stage(0, 0);

  for (int it = 0; it < N_SP / 32; ++it) {
    wait_async0();         // my share of chunk `it` copies landed
    __syncthreads();       // everyone's share landed; prev buf free for reuse
    if (it + 1 < N_SP / 32) stage((it + 1) * 32, (it + 1) & 1);

    const int buf = it & 1;
    const bf16* KsB = Ks + buf * KS_BUF;
    const bf16* VsB = Vs + buf * VS_BUF;
    const bf16* k0 = KsB + col * KS_STRIDE;          // keys m0..m0+15
    const bf16* k1 = KsB + (col + 16) * KS_STRIDE;   // keys m0+16..m0+31

    union { v16bf v; v8bf h[2]; } A;
    v8f s0, s1;
#pragma unroll
    for (int r = 0; r < 8; ++r) { s0[r] = 0.f; s1[r] = 0.f; }

    A.h[0] = *reinterpret_cast<const v8bf*>(k0 + half * 16);
    A.h[1] = *reinterpret_cast<const v8bf*>(k0 + half * 16 + 8);
    s0 = wmma_bf16(A.v, bq0, s0);
    A.h[0] = *reinterpret_cast<const v8bf*>(k0 + 32 + half * 16);
    A.h[1] = *reinterpret_cast<const v8bf*>(k0 + 40 + half * 16);
    s0 = wmma_bf16(A.v, bq1, s0);
    A.h[0] = *reinterpret_cast<const v8bf*>(k1 + half * 16);
    A.h[1] = *reinterpret_cast<const v8bf*>(k1 + half * 16 + 8);
    s1 = wmma_bf16(A.v, bq0, s1);
    A.h[0] = *reinterpret_cast<const v8bf*>(k1 + 32 + half * 16);
    A.h[1] = *reinterpret_cast<const v8bf*>(k1 + 40 + half * 16);
    s1 = wmma_bf16(A.v, bq1, s1);

    // ---- online softmax (per query n = lane column) ----
    float cm = s0[0];
#pragma unroll
    for (int r = 0; r < 8; ++r) { cm = fmaxf(cm, s0[r]); cm = fmaxf(cm, s1[r]); }
    cm = fmaxf(cm, swz16f(cm));
    const float mn = fmaxf(mi, cm);

    // Rescale only when some lane's max actually moved (wave-uniform branch).
    if (__builtin_amdgcn_ballot_w32(mn > mi) != 0u) {
      const float scale = __expf(mi - mn);
      li *= scale;
#pragma unroll
      for (int t = 0; t < 16; ++t)
#pragma unroll
        for (int r = 0; r < 8; ++r) acc[t][r] *= scale;
    }
    mi = mn;

    float p0[8], p1[8], ls = 0.f;
#pragma unroll
    for (int r = 0; r < 8; ++r) {
      p0[r] = __expf(s0[r] - mn);
      p1[r] = __expf(s1[r] - mn);
      ls += p0[r] + p1[r];
    }
    li += ls + swz16f(ls);

    // ---- build P as PV B-operand (32 keys x 16 queries), one SWAPX16 ----
    union { v8bf v; int i[4]; } own, rec;
#pragma unroll
    for (int r = 0; r < 8; ++r) own.v[r] = (bf16)(half ? p0[r] : p1[r]);
#pragma unroll
    for (int j = 0; j < 4; ++j) rec.i[j] = swz16i(own.i[j]);
    union { v16bf v; v8bf h[2]; } P;
#pragma unroll
    for (int r = 0; r < 8; ++r) {
      P.h[0][r] = half ? rec.v[r] : (bf16)p0[r];
      P.h[1][r] = half ? (bf16)p1[r] : rec.v[r];
    }

    // ---- PV: acc[t](16d x 16n) += V^T tile (16d x 32m) * P (32m x 16n) ----
#pragma unroll
    for (int t = 0; t < 16; ++t) {
      const bf16* vr = VsB + (t * 16 + col) * VS_STRIDE;
      union { v16bf v; v8bf h[2]; } Av;
      Av.h[0] = *reinterpret_cast<const v8bf*>(vr + half * 16);
      Av.h[1] = *reinterpret_cast<const v8bf*>(vr + half * 16 + 8);
      acc[t] = wmma_bf16(Av.v, P.v, acc[t]);
    }
  }

  const float inv = 1.0f / li;
  float* ob = out + (size_t)b * OUT_CH * N_SP + n0 + col;
#pragma unroll
  for (int t = 0; t < 16; ++t)
#pragma unroll
    for (int r = 0; r < 8; ++r)
      ob[(size_t)(t * 16 + half * 8 + r) * N_SP] = acc[t][r] * inv;
}

// ---------------------------------------------------------------------------
extern "C" void kernel_launch(void* const* d_in, const int* in_sizes, int n_in,
                              void* d_out, int out_size, void* d_ws, size_t ws_size,
                              hipStream_t stream) {
  (void)in_sizes; (void)n_in; (void)out_size; (void)ws_size;
  const float* x  = (const float*)d_in[0];
  const float* wq = (const float*)d_in[1];
  const float* bq = (const float*)d_in[2];
  const float* wk = (const float*)d_in[3];
  const float* bk = (const float*)d_in[4];
  const float* wv = (const float*)d_in[5];
  const float* bv = (const float*)d_in[6];
  float* out = (float*)d_out;

  bf16* Qw = (bf16*)d_ws;                               // [B,4096,64]  4 MB
  bf16* Kw = Qw + (size_t)BATCH * N_SP * QK;            // [B,4096,64]  4 MB
  bf16* Vt = Kw + (size_t)BATCH * N_SP * QK;            // [B,256,4096] 16.8 MB

  proj_kernel<<<dim3(32, BATCH), 256, 0, stream>>>(x, wq, bq, wk, bk, wv, bv,
                                                   Qw, Kw, Vt);
  attn_kernel<<<dim3(N_SP / 64, BATCH), 128, 0, stream>>>(Qw, Kw, Vt, out);
}
